// DeepSC_15908558864395
// MI455X (gfx1250) — compile-verified, matrix operand order checked by reading, not run on hardware
//
#include <hip/hip_runtime.h>
#include <hip/hip_bf16.h>
#include <math.h>

#define LAYERS  4
#define DMODEL  256
#define NHEAD   8
#define HDIM    32
#define GLEN    512
#define BATCH   2
#define NTOK    (BATCH*GLEN)      // 1024
#define DFF     (4*DMODEL)        // 1024
#define NBH     (BATCH*NHEAD)     // 16

typedef __attribute__((ext_vector_type(16))) _Float16 v16h;
typedef __attribute__((ext_vector_type(8)))  _Float16 v8h;
typedef __attribute__((ext_vector_type(8)))  float    v8f;

union HFrag { v16h v; v8h h[2]; };

// Load a 16x32 f16 WMMA fragment (A layout, or B^T stored N-major/K-contiguous).
// Per lane: row = lane%16, k-halves {kh..kh+7, kh+16..kh+23}, kh = (lane/16)*8
// => two contiguous 16-byte loads.
__device__ __forceinline__ v16h load_frag16(const _Float16* __restrict__ base, int ld) {
  int lane = threadIdx.x & 31;
  int row  = lane & 15;
  int kh   = (lane >> 4) << 3;
  const _Float16* p = base + (size_t)row * ld + kh;
  HFrag f;
  f.h[0] = *(const v8h*)(p);
  f.h[1] = *(const v8h*)(p + 16);
  return f.v;
}

#define WMMA_F16(A, B, Cacc) \
  __builtin_amdgcn_wmma_f32_16x16x32_f16(false, (A), false, (B), (short)0, (Cacc), false, false)

__device__ __forceinline__ float wred_sum(float v) {
  #pragma unroll
  for (int o = 16; o > 0; o >>= 1) v += __shfl_xor(v, o, 32);
  return v;
}
__device__ __forceinline__ float wred_max(float v) {
  #pragma unroll
  for (int o = 16; o > 0; o >>= 1) v = fmaxf(v, __shfl_xor(v, o, 32));
  return v;
}

// ---------------- Generic WMMA GEMM: C[N,M] = epi(A@W + bias + R1 + R2) ------
// 32x32 output per wave (2x2 WMMA tiles, fragments reused twice each).
// A1:(NTOK,k1) f16 row-major; optional A2:(NTOK,k2) continues the K loop.
// Wt:(M,K) f16 = W^T (N-major, K-contiguous), kw0 = starting K column in Wt.
// epi: 0 = none, 1 = exact GELU.  C (f32) and C16 (f16, scaled) optional.
__global__ __launch_bounds__(256) void gemm16(
    const _Float16* __restrict__ A1, int lda1,
    const _Float16* __restrict__ A2, int lda2,
    const _Float16* __restrict__ Wt, int ldw, int kw0, int k1, int k2,
    const float* __restrict__ bias,
    const float* __restrict__ R1, const float* __restrict__ R2,
    float* __restrict__ C, _Float16* __restrict__ C16,
    int Mc, float hscale, int epi)
{
  int wave    = (blockIdx.x * blockDim.x + threadIdx.x) >> 5;
  int stripsM = Mc >> 5;
  int wm = wave / stripsM;
  int wn = wave - wm * stripsM;
  int lane = threadIdx.x & 31;

  v8f acc[2][2] = {};
  const _Float16* a = A1 + (size_t)wm * 32 * lda1;
  const _Float16* w = Wt + (size_t)wn * 32 * ldw + kw0;
  for (int k = 0; k < k1; k += 32) {
    __builtin_prefetch(a + k + 256, 0, 1);
    __builtin_prefetch(w + k + 256, 0, 1);
    v16h a0 = load_frag16(a + k, lda1);
    v16h a1 = load_frag16(a + (size_t)16 * lda1 + k, lda1);
    v16h b0 = load_frag16(w + k, ldw);
    v16h b1 = load_frag16(w + (size_t)16 * ldw + k, ldw);
    acc[0][0] = WMMA_F16(a0, b0, acc[0][0]);
    acc[0][1] = WMMA_F16(a0, b1, acc[0][1]);
    acc[1][0] = WMMA_F16(a1, b0, acc[1][0]);
    acc[1][1] = WMMA_F16(a1, b1, acc[1][1]);
  }
  if (A2) {
    const _Float16* a2 = A2 + (size_t)wm * 32 * lda2;
    for (int k = 0; k < k2; k += 32) {
      v16h a0 = load_frag16(a2 + k, lda2);
      v16h a1 = load_frag16(a2 + (size_t)16 * lda2 + k, lda2);
      v16h b0 = load_frag16(w + k1 + k, ldw);
      v16h b1 = load_frag16(w + (size_t)16 * ldw + k1 + k, ldw);
      acc[0][0] = WMMA_F16(a0, b0, acc[0][0]);
      acc[0][1] = WMMA_F16(a0, b1, acc[0][1]);
      acc[1][0] = WMMA_F16(a1, b0, acc[1][0]);
      acc[1][1] = WMMA_F16(a1, b1, acc[1][1]);
    }
  }
  int cn = lane & 15;
  int rb = (lane >> 4) << 3;
  #pragma unroll
  for (int ti = 0; ti < 2; ++ti) {
    #pragma unroll
    for (int tj = 0; tj < 2; ++tj) {
      int col = wn * 32 + tj * 16 + cn;
      float bv = bias ? bias[col] : 0.f;
      #pragma unroll
      for (int v = 0; v < 8; ++v) {
        int r = wm * 32 + ti * 16 + rb + v;
        size_t idx = (size_t)r * Mc + col;
        float x = acc[ti][tj][v] + bv;
        if (R1) x += R1[idx];
        if (R2) x += R2[idx];
        if (epi == 1) x = 0.5f * x * (1.0f + erff(x * 0.70710678118654752f));
        if (C)   C[idx]   = x;
        if (C16) C16[idx] = (_Float16)(x * hscale);
      }
    }
  }
}

// ---------------- Attention scores: S[bh,q,kk] = Qh . Kh (Q pre-scaled) ------
// 32x32 scores per wave.
__global__ __launch_bounds__(256) void attn_scores(
    const _Float16* __restrict__ Q, const _Float16* __restrict__ K,
    float* __restrict__ S)
{
  int wave = (blockIdx.x * blockDim.x + threadIdx.x) >> 5;
  int bh   = wave >> 8;           // (G/32)^2 = 256 tiles per (b,h)
  int tile = wave & 255;
  int tq = tile >> 4, tk = tile & 15;
  int b = bh >> 3, h = bh & 7;
  const _Float16* qb = Q + (size_t)b * GLEN * DMODEL + h * HDIM + (size_t)tq * 32 * DMODEL;
  const _Float16* kb = K + (size_t)b * GLEN * DMODEL + h * HDIM + (size_t)tk * 32 * DMODEL;
  v16h a0 = load_frag16(qb, DMODEL);
  v16h a1 = load_frag16(qb + (size_t)16 * DMODEL, DMODEL);
  v16h b0 = load_frag16(kb, DMODEL);
  v16h b1 = load_frag16(kb + (size_t)16 * DMODEL, DMODEL);
  v8f acc[2][2] = {};
  acc[0][0] = WMMA_F16(a0, b0, acc[0][0]);
  acc[0][1] = WMMA_F16(a0, b1, acc[0][1]);
  acc[1][0] = WMMA_F16(a1, b0, acc[1][0]);
  acc[1][1] = WMMA_F16(a1, b1, acc[1][1]);
  int lane = threadIdx.x & 31;
  int cn = lane & 15, rb = (lane >> 4) << 3;
  float* s = S + (size_t)bh * GLEN * GLEN;
  #pragma unroll
  for (int ti = 0; ti < 2; ++ti)
    #pragma unroll
    for (int tj = 0; tj < 2; ++tj)
      #pragma unroll
      for (int v = 0; v < 8; ++v)
        s[(size_t)(tq * 32 + ti * 16 + rb + v) * GLEN + tk * 32 + tj * 16 + cn] =
            acc[ti][tj][v];
}

// ---------------- Abar @ V : O16[b,g, h*32+dh] -------------------------------
// 32 rows x full 32-wide head per wave (2x2 tiles).
__global__ __launch_bounds__(256) void attn_av(
    const _Float16* __restrict__ P, const _Float16* __restrict__ Vt,
    _Float16* __restrict__ O16)
{
  int wave = (blockIdx.x * blockDim.x + threadIdx.x) >> 5;
  int bh   = wave >> 4;           // G/32 = 16 row-strips per (b,h)
  int tq   = wave & 15;
  int b = bh >> 3, h = bh & 7;
  const _Float16* pa = P  + (size_t)bh * GLEN * GLEN + (size_t)tq * 32 * GLEN;
  const _Float16* vb = Vt + (size_t)bh * HDIM * GLEN;
  v8f acc[2][2] = {};
  for (int k = 0; k < GLEN; k += 32) {
    v16h a0 = load_frag16(pa + k, GLEN);
    v16h a1 = load_frag16(pa + (size_t)16 * GLEN + k, GLEN);
    v16h b0 = load_frag16(vb + k, GLEN);
    v16h b1 = load_frag16(vb + (size_t)16 * GLEN + k, GLEN);
    acc[0][0] = WMMA_F16(a0, b0, acc[0][0]);
    acc[0][1] = WMMA_F16(a0, b1, acc[0][1]);
    acc[1][0] = WMMA_F16(a1, b0, acc[1][0]);
    acc[1][1] = WMMA_F16(a1, b1, acc[1][1]);
  }
  int lane = threadIdx.x & 31;
  int cn = lane & 15, rb = (lane >> 4) << 3;
  #pragma unroll
  for (int ti = 0; ti < 2; ++ti)
    #pragma unroll
    for (int tj = 0; tj < 2; ++tj)
      #pragma unroll
      for (int v = 0; v < 8; ++v) {
        size_t g = (size_t)b * GLEN + tq * 32 + ti * 16 + rb + v;
        O16[g * DMODEL + h * HDIM + tj * 16 + cn] = (_Float16)acc[ti][tj][v];
      }
}

// ---------------- softmax (+ optional mask renorm), one wave per row ---------
__global__ __launch_bounds__(256) void softmax_mask(
    const float* __restrict__ S, const float* __restrict__ Msk,
    _Float16* __restrict__ P)
{
  int wave = (blockIdx.x * 256 + threadIdx.x) >> 5;
  int lane = threadIdx.x & 31;
  int bh = wave >> 9;
  int q  = wave & 511;
  int b  = bh >> 3;
  const float* row = S + (size_t)bh * GLEN * GLEN + (size_t)q * GLEN;
  float vals[16];
  float mx = -3.4e38f;
  #pragma unroll
  for (int j = 0; j < 16; ++j) { vals[j] = row[lane + j * 32]; mx = fmaxf(mx, vals[j]); }
  mx = wred_max(mx);
  float sum = 0.f;
  #pragma unroll
  for (int j = 0; j < 16; ++j) { vals[j] = expf(vals[j] - mx); sum += vals[j]; }
  sum = wred_sum(sum);
  float inv = 1.0f / sum;
  _Float16* out = P + (size_t)bh * GLEN * GLEN + (size_t)q * GLEN;
  if (Msk) {
    const float* mrow = Msk + ((size_t)(b * GLEN + q)) * GLEN;
    float t[16]; float s2 = 0.f;
    #pragma unroll
    for (int j = 0; j < 16; ++j) {
      t[j] = vals[j] * inv * mrow[lane + j * 32];
      s2 += fabsf(t[j]);
    }
    s2 = wred_sum(s2);
    float d = 1.0f / (s2 + 1e-8f);
    #pragma unroll
    for (int j = 0; j < 16; ++j) out[lane + j * 32] = (_Float16)(t[j] * d);
  } else {
    #pragma unroll
    for (int j = 0; j < 16; ++j) out[lane + j * 32] = (_Float16)(vals[j] * inv);
  }
}

// ---------------- LayerNorm, one wave per row (D=256) ------------------------
__global__ __launch_bounds__(256) void layernorm(
    const float* __restrict__ X, const float* __restrict__ w,
    const float* __restrict__ bb, float* __restrict__ Y,
    _Float16* __restrict__ Y16)
{
  int wave = (blockIdx.x * 256 + threadIdx.x) >> 5;
  int lane = threadIdx.x & 31;
  const float* x = X + (size_t)wave * DMODEL;
  float v[8]; float s = 0.f;
  #pragma unroll
  for (int j = 0; j < 8; ++j) { v[j] = x[lane + j * 32]; s += v[j]; }
  float mean = wred_sum(s) * (1.0f / DMODEL);
  float var = 0.f;
  #pragma unroll
  for (int j = 0; j < 8; ++j) { float d = v[j] - mean; var += d * d; }
  var = wred_sum(var) * (1.0f / DMODEL);
  float r = rsqrtf(var + 1e-5f);
  #pragma unroll
  for (int j = 0; j < 8; ++j) {
    int c = lane + j * 32;
    float o = (v[j] - mean) * r * w[c] + bb[c];
    if (Y)   Y[(size_t)wave * DMODEL + c]   = o;
    if (Y16) Y16[(size_t)wave * DMODEL + c] = (_Float16)o;
  }
}

// ---------------- Weight transpose+convert: Wt[m,k] = f16(W[k,m]) ------------
__global__ void wconv(const float* __restrict__ W, _Float16* __restrict__ Wt,
                      int K, int M)
{
  int idx = blockIdx.x * 256 + threadIdx.x;
  if (idx >= K * M) return;
  int m = idx / K, k = idx - m * K;
  Wt[idx] = (_Float16)W[(size_t)k * M + m];
}

// ---------------- V (b,g,D) f16 -> Vt (b,h,hd,G) f16 -------------------------
__global__ void v_transpose(const _Float16* __restrict__ V16,
                            _Float16* __restrict__ Vt)
{
  int idx = blockIdx.x * 256 + threadIdx.x;
  if (idx >= NTOK * DMODEL) return;
  int b = idx / (GLEN * DMODEL);
  int rem = idx - b * GLEN * DMODEL;
  int g = rem / DMODEL;
  int d = rem - g * DMODEL;
  int h = d >> 5, dh = d & 31;
  Vt[(((size_t)b * NHEAD + h) * HDIM + dh) * GLEN + g] = V16[idx];
}

// ---------------- Embeddings -------------------------------------------------
__global__ void embed(const int* __restrict__ gid, const int* __restrict__ did,
                      const float* __restrict__ nex,
                      const float* __restrict__ gtab, const float* __restrict__ btab,
                      const float* __restrict__ cproj,
                      float* __restrict__ g32, _Float16* __restrict__ g16,
                      float* __restrict__ e32, _Float16* __restrict__ e16)
{
  int idx = blockIdx.x * 256 + threadIdx.x;
  if (idx >= NTOK * DMODEL) return;
  int t = idx >> 8, d = idx & 255;
  float gv = gtab[(size_t)gid[t] * DMODEL + d];
  float ev = btab[(size_t)did[t] * DMODEL + d] + 0.1f * nex[t] * cproj[d];
  g32[idx] = gv; g16[idx] = (_Float16)gv;
  e32[idx] = ev; e16[idx] = (_Float16)ev;
}

// ---------------- Straight-through Gumbel mask: M in {-1,0,+1} ---------------
// Block = 16 query rows; 16x16 (q,k) tile per pass with src/tgt/Wg2 in LDS.
// Cuts tgt re-reads 16x vs. one-row-per-block.
__global__ __launch_bounds__(256) void gumbel_mask(
    const float* __restrict__ src, const float* __restrict__ tgt,
    const float* __restrict__ Wg2, const float* __restrict__ bg2,
    const float* __restrict__ U, float* __restrict__ Mout)
{
  __shared__ float s[16][DMODEL + 1];
  __shared__ float t[16][DMODEL + 1];
  __shared__ float w2s[DMODEL * 3];
  int bq0 = blockIdx.x * 16;          // global (b*G + q) base; never crosses b
  int b   = bq0 >> 9;
  int tid = threadIdx.x;
  for (int i = tid; i < 16 * DMODEL; i += 256)
    s[i >> 8][i & 255] = src[(size_t)(bq0 + (i >> 8)) * DMODEL + (i & 255)];
  for (int i = tid; i < 3 * DMODEL; i += 256) w2s[i] = Wg2[i];
  float b0 = bg2[0], b1 = bg2[1], b2 = bg2[2];
  int qi = tid >> 4;                  // 0..15
  int kl = tid & 15;                  // 0..15
  for (int kk0 = 0; kk0 < GLEN; kk0 += 16) {
    __syncthreads();
    for (int i = tid; i < 16 * DMODEL; i += 256)
      t[i >> 8][i & 255] = tgt[(size_t)(b * GLEN + kk0 + (i >> 8)) * DMODEL + (i & 255)];
    __syncthreads();
    float a0 = 0.f, a1 = 0.f, a2 = 0.f;
    for (int d = 0; d < DMODEL; ++d) {
      float h = fmaxf(s[qi][d] + t[kl][d], 0.f);
      a0 += h * w2s[d * 3 + 0];
      a1 += h * w2s[d * 3 + 1];
      a2 += h * w2s[d * 3 + 2];
    }
    int kk = kk0 + kl;
    const float* u = U + ((size_t)(bq0 + qi) * GLEN + kk) * 3;
    float g0 = -logf(-logf(u[0] + 1e-20f) + 1e-20f);
    float g1 = -logf(-logf(u[1] + 1e-20f) + 1e-20f);
    float g2 = -logf(-logf(u[2] + 1e-20f) + 1e-20f);
    float l0 = a0 + b0 + g0, l1 = a1 + b1 + g1, l2 = a2 + b2 + g2;
    int arg = 0; float best = l0;
    if (l1 > best) { best = l1; arg = 1; }
    if (l2 > best) { best = l2; arg = 2; }
    Mout[(size_t)(bq0 + qi) * GLEN + kk] =
        (arg == 2 ? 1.f : 0.f) - (arg == 0 ? 1.f : 0.f);
  }
}

__global__ void copy_out(const float* __restrict__ g, const float* __restrict__ e,
                         float* __restrict__ out)
{
  int i = blockIdx.x * 256 + threadIdx.x;
  if (i < NTOK * DMODEL) { out[i] = g[i]; out[NTOK * DMODEL + i] = e[i]; }
}

// ============================= host side =====================================
enum {
  pWv_g, pWv_e, pWq_g, pWk_g, pWo_g, pWq_e, pWk_e, pWo_e,
  pbv_g, pbv_e, pbq_g, pbk_g, pbo_g, pbq_e, pbk_e, pbo_e, pbfuse, pbf2_g, pbf2_e,
  pWfuse, pWf1_g, pWf1_e, pbf1_g, pbf1_e, pWf2_g, pWf2_e,
  pln_g1_w, pln_g1_b, pln_g2_w, pln_g2_b, pln_e1_w, pln_e1_b, pln_e2_w, pln_e2_b,
  pWg1, pbg1, pWg2, pbg2, pgene_table, pbin_table, pcont_proj, NPARAM
};

extern "C" void kernel_launch(void* const* d_in, const int* in_sizes, int n_in,
                              void* d_out, int out_size, void* d_ws, size_t ws_size,
                              hipStream_t stream) {
  (void)in_sizes; (void)n_in; (void)out_size; (void)ws_size;
  const int*   gid = (const int*)d_in[0];
  const int*   did = (const int*)d_in[1];
  const float* nex = (const float*)d_in[2];
  const float* gum = (const float*)d_in[3];
  const float* prm[NPARAM];
  for (int i = 0; i < NPARAM; ++i) prm[i] = (const float*)d_in[4 + i];

  // ---- bump allocator on workspace ----
  char* wp = (char*)d_ws;
  auto alloc = [&](size_t bytes) -> char* {
    char* r = wp; wp += (bytes + 255) & ~(size_t)255; return r;
  };
  auto fal = [&](size_t n) -> float*    { return (float*)alloc(n * sizeof(float)); };
  auto hal = [&](size_t n) -> _Float16* { return (_Float16*)alloc(n * sizeof(_Float16)); };

  // transposed f16 weights
  _Float16* wdd[LAYERS][8];
  _Float16 *wfu[LAYERS], *wf1g[LAYERS], *wf1e[LAYERS], *wf2g[LAYERS], *wf2e[LAYERS];
  for (int i = 0; i < LAYERS; ++i) {
    for (int j = 0; j < 8; ++j) wdd[i][j] = hal((size_t)DMODEL * DMODEL);
    wfu[i]  = hal((size_t)2 * DMODEL * DMODEL);
    wf1g[i] = hal((size_t)DMODEL * DFF);
    wf1e[i] = hal((size_t)DMODEL * DFF);
    wf2g[i] = hal((size_t)DFF * DMODEL);
    wf2e[i] = hal((size_t)DFF * DMODEL);
  }
  _Float16* wg1t = hal((size_t)2 * DMODEL * DMODEL);

  // activations
  const size_t ND = (size_t)NTOK * DMODEL;
  float *gA32 = fal(ND), *gB32 = fal(ND), *eA32 = fal(ND), *eB32 = fal(ND);
  _Float16 *gA16 = hal(ND), *gB16 = hal(ND), *eA16 = hal(ND), *eB16 = hal(ND);
  float* x1_32   = fal(ND);  _Float16* x1_16   = hal(ND);
  _Float16* vbuf16 = hal(ND);
  _Float16* vt16   = hal((size_t)NBH * HDIM * GLEN);
  _Float16 *q16 = hal(ND), *k16 = hal(ND);
  float*     Sbuf = fal((size_t)NBH * GLEN * GLEN);
  _Float16*  Pbuf = hal((size_t)NBH * GLEN * GLEN);
  _Float16* attn16 = hal(ND);
  float* x2_32   = fal(ND);
  float* xln2_32 = fal(ND);  _Float16* xln2_16 = hal(ND);
  _Float16* h16    = hal((size_t)NTOK * DFF);
  _Float16* fused16 = hal(ND);
  float* src32 = fal(ND);
  float* tgt32 = fal(ND);
  float* Mbuf  = fal((size_t)BATCH * GLEN * GLEN);

  auto conv = [&](const float* W, _Float16* Wt, int K, int M) {
    int tot = K * M;
    wconv<<<(tot + 255) / 256, 256, 0, stream>>>(W, Wt, K, M);
  };
  auto gemm = [&](const _Float16* A1, int lda1, const _Float16* A2, int lda2,
                  const _Float16* Wt, int ldw, int kw0, int k1, int k2,
                  const float* bias, const float* R1, const float* R2,
                  float* C, _Float16* C16, int Mc, float hs, int epi) {
    int waves = (NTOK / 32) * (Mc / 32);     // 32x32 strip per wave
    gemm16<<<waves / 8, 256, 0, stream>>>(A1, lda1, A2, lda2, Wt, ldw, kw0, k1, k2,
                                          bias, R1, R2, C, C16, Mc, hs, epi);
  };
  auto lnorm = [&](const float* X, const float* w, const float* b,
                   float* Y, _Float16* Y16) {
    layernorm<<<NTOK / 8, 256, 0, stream>>>(X, w, b, Y, Y16);
  };

  // convert all weights (deterministic each call)
  for (int i = 0; i < LAYERS; ++i) {
    for (int j = 0; j < 8; ++j)
      conv(prm[j] + (size_t)i * DMODEL * DMODEL, wdd[i][j], DMODEL, DMODEL);
    conv(prm[pWfuse] + (size_t)i * 2 * DMODEL * DMODEL, wfu[i], 2 * DMODEL, DMODEL);
    conv(prm[pWf1_g] + (size_t)i * DMODEL * DFF, wf1g[i], DMODEL, DFF);
    conv(prm[pWf1_e] + (size_t)i * DMODEL * DFF, wf1e[i], DMODEL, DFF);
    conv(prm[pWf2_g] + (size_t)i * DFF * DMODEL, wf2g[i], DFF, DMODEL);
    conv(prm[pWf2_e] + (size_t)i * DFF * DMODEL, wf2e[i], DFF, DMODEL);
  }
  conv(prm[pWg1], wg1t, 2 * DMODEL, DMODEL);

  const float QSCALE = 0.17677669529663687f;  // 1/sqrt(32)

  float *cg32 = gA32, *ce32 = eA32, *ng32 = gB32, *ne32 = eB32;
  _Float16 *cg16 = gA16, *ce16 = eA16, *ng16 = gB16, *ne16 = eB16;

  embed<<<(NTOK * DMODEL + 255) / 256, 256, 0, stream>>>(
      gid, did, nex, prm[pgene_table], prm[pbin_table], prm[pcont_proj],
      cg32, cg16, ce32, ce16);

  for (int i = 0; i < LAYERS; ++i) {
    const size_t iD = (size_t)i * DMODEL;
    const size_t iF = (size_t)i * DFF;
    const float* mptr = nullptr;
    if (i >= 2) {
      gemm(cg16, DMODEL, nullptr, 0, wg1t, 2 * DMODEL, 0, DMODEL, 0,
           prm[pbg1], nullptr, nullptr, src32, nullptr, DMODEL, 1.f, 0);
      gemm(cg16, DMODEL, nullptr, 0, wg1t, 2 * DMODEL, DMODEL, DMODEL, 0,
           nullptr, nullptr, nullptr, tgt32, nullptr, DMODEL, 1.f, 0);
      gumbel_mask<<<NTOK / 16, 256, 0, stream>>>(
          src32, tgt32, prm[pWg2], prm[pbg2],
          gum + (size_t)(i - 2) * BATCH * GLEN * GLEN * 3, Mbuf);
      mptr = Mbuf;
    }
    // ---- gene branch ----
    lnorm(cg32, prm[pln_g1_w] + iD, prm[pln_g1_b] + iD, x1_32, x1_16);
    gemm(x1_16, DMODEL, nullptr, 0, wdd[i][pWv_g], DMODEL, 0, DMODEL, 0,
         prm[pbv_g] + iD, nullptr, nullptr, nullptr, vbuf16, DMODEL, 1.f, 0);
    v_transpose<<<(NTOK * DMODEL + 255) / 256, 256, 0, stream>>>(vbuf16, vt16);
    gemm(x1_16, DMODEL, nullptr, 0, wdd[i][pWq_g], DMODEL, 0, DMODEL, 0,
         prm[pbq_g] + iD, nullptr, nullptr, nullptr, q16, DMODEL, QSCALE, 0);
    gemm(x1_16, DMODEL, nullptr, 0, wdd[i][pWk_g], DMODEL, 0, DMODEL, 0,
         prm[pbk_g] + iD, nullptr, nullptr, nullptr, k16, DMODEL, 1.f, 0);
    attn_scores<<<512, 256, 0, stream>>>(q16, k16, Sbuf);
    softmax_mask<<<1024, 256, 0, stream>>>(Sbuf, mptr, Pbuf);
    attn_av<<<32, 256, 0, stream>>>(Pbuf, vt16, attn16);
    gemm(attn16, DMODEL, nullptr, 0, wdd[i][pWo_g], DMODEL, 0, DMODEL, 0,
         prm[pbo_g] + iD, cg32, nullptr, x2_32, nullptr, DMODEL, 1.f, 0);
    lnorm(x2_32, prm[pln_g2_w] + iD, prm[pln_g2_b] + iD, xln2_32, xln2_16);
    gemm(xln2_16, DMODEL, nullptr, 0, wf1g[i], DMODEL, 0, DMODEL, 0,
         prm[pbf1_g] + iF, nullptr, nullptr, nullptr, h16, DFF, 1.f, 1);
    gemm(h16, DFF, nullptr, 0, wf2g[i], DFF, 0, DFF, 0,
         prm[pbf2_g] + iD, x2_32, xln2_32, ng32, ng16, DMODEL, 1.f, 0);
    // ---- expression branch (reads pre-update cg16/ce16) ----
    lnorm(ce32, prm[pln_e1_w] + iD, prm[pln_e1_b] + iD, x1_32, x1_16);
    gemm(x1_16, DMODEL, nullptr, 0, wdd[i][pWv_e], DMODEL, 0, DMODEL, 0,
         prm[pbv_e] + iD, nullptr, nullptr, nullptr, vbuf16, DMODEL, 1.f, 0);
    v_transpose<<<(NTOK * DMODEL + 255) / 256, 256, 0, stream>>>(vbuf16, vt16);
    gemm(cg16, DMODEL, ce16, DMODEL, wfu[i], 2 * DMODEL, 0, DMODEL, DMODEL,
         prm[pbfuse] + iD, nullptr, nullptr, nullptr, fused16, DMODEL, 1.f, 0);
    gemm(fused16, DMODEL, nullptr, 0, wdd[i][pWq_e], DMODEL, 0, DMODEL, 0,
         prm[pbq_e] + iD, nullptr, nullptr, nullptr, q16, DMODEL, QSCALE, 0);
    gemm(fused16, DMODEL, nullptr, 0, wdd[i][pWk_e], DMODEL, 0, DMODEL, 0,
         prm[pbk_e] + iD, nullptr, nullptr, nullptr, k16, DMODEL, 1.f, 0);
    attn_scores<<<512, 256, 0, stream>>>(q16, k16, Sbuf);
    softmax_mask<<<1024, 256, 0, stream>>>(Sbuf, mptr, Pbuf);
    attn_av<<<32, 256, 0, stream>>>(Pbuf, vt16, attn16);
    gemm(attn16, DMODEL, nullptr, 0, wdd[i][pWo_e], DMODEL, 0, DMODEL, 0,
         prm[pbo_e] + iD, ce32, nullptr, x2_32, nullptr, DMODEL, 1.f, 0);
    lnorm(x2_32, prm[pln_e2_w] + iD, prm[pln_e2_b] + iD, xln2_32, xln2_16);
    gemm(xln2_16, DMODEL, nullptr, 0, wf1e[i], DMODEL, 0, DMODEL, 0,
         prm[pbf1_e] + iF, nullptr, nullptr, nullptr, h16, DFF, 1.f, 1);
    gemm(h16, DFF, nullptr, 0, wf2e[i], DFF, 0, DFF, 0,
         prm[pbf2_e] + iD, x2_32, xln2_32, ne32, ne16, DMODEL, 1.f, 0);
    // ping-pong
    { float* t = cg32; cg32 = ng32; ng32 = t; }
    { float* t = ce32; ce32 = ne32; ne32 = t; }
    { _Float16* t = cg16; cg16 = ng16; ng16 = t; }
    { _Float16* t = ce16; ce16 = ne16; ne16 = t; }
  }

  copy_out<<<(NTOK * DMODEL + 255) / 256, 256, 0, stream>>>(cg32, ce32, (float*)d_out);
}
